// StructureModule_52286931861852
// MI455X (gfx1250) — compile-verified, matrix operand order checked by reading, not run on hardware
//
#include <hip/hip_runtime.h>
#include <hip/hip_bf16.h>

typedef __attribute__((ext_vector_type(16))) _Float16 v16h;
typedef __attribute__((ext_vector_type(8)))  _Float16 v8h;
typedef __attribute__((ext_vector_type(8)))  float    v8f;

#define DEVFN static __device__ __forceinline__

// ---- problem constants ----
#define NR   768      // residues
#define CS   384
#define CZ   128
#define NH   12
#define CH   16
#define PQ   4
#define PV   8
#define CANG 128
#define SCALE_ 10.0f
#define INF_  100000.0f

// ---------------------------------------------------------------------------
// WMMA fragment helpers (wave32, v_wmma_f32_16x16x32_f16)
// A 16x32 f16: lane m=lane&15, halves = cols [kh,kh+8) and [16+kh,16+kh+8),
// kh = (lane>>4)*8.  B supplied K-major (Bt[n][k] row-major) => same loader.
// ---------------------------------------------------------------------------
DEVFN v16h frag_ld(const _Float16* base, int stride_h) {
  int lane = threadIdx.x & 31;
  const _Float16* r = base + (size_t)(lane & 15) * stride_h + ((lane >> 4) << 3);
  union { v16h v; v8h h[2]; } u;
  u.h[0] = *(const v8h*)(r);
  u.h[1] = *(const v8h*)(r + 16);
  return u.v;
}
DEVFN v8f wmma_f16(v16h a, v16h b, v8f c) {
  return __builtin_amdgcn_wmma_f32_16x16x32_f16(false, a, false, b, (short)0, c,
                                                false, false);
}

// ---------------------------------------------------------------------------
// Weight pack: W (K x N, f32 row-major) -> Wt (Npad x K, f16), zero padded.
// Done once per launch; weights stay f16 + pre-transposed => GEMM B operand
// is loaded directly from global as b128 fragments (L2-resident, ~4MB total).
// ---------------------------------------------------------------------------
__global__ void pack_w_kernel(const float* __restrict__ W, _Float16* __restrict__ Wt,
                              int K, int N, int Npad)
{
  long idx = (long)blockIdx.x * blockDim.x + threadIdx.x;
  if (idx >= (long)Npad * K) return;
  int k = (int)(idx / Npad), n = (int)(idx % Npad);
  Wt[(long)n * K + k] = (n < N) ? (_Float16)W[(long)k * N + n] : (_Float16)0.f;
}

// ---------------------------------------------------------------------------
// Generic WMMA GEMM: C[M,N] = act(scale*(A[M,K] @ B + bias)) (+ res)
// A: f32 / f16 / zT-indexed f16 (compile-time). B: packed f16 [Npad>=tile][K].
// M%32==0, K%32==0. Block = 256 thr (8 waves), tile 32x64.
// Only A is staged through LDS; B fragments load straight from global.
// ---------------------------------------------------------------------------
#define GF_RELU_A 1
#define GF_RELU_O 2
#define GF_TRANS  4
#define GF_A_F16  8
#define GF_A_ZT   16

template <int FLAGS>
__global__ __launch_bounds__(256)
void gemm_wmma_kernel(const void* __restrict__ Av, const _Float16* __restrict__ Bt,
                      const float* __restrict__ bias, const float* __restrict__ res,
                      float* __restrict__ C,
                      int M, int N, int K, float out_scale, int ldc,
                      long aB, long bBh, long cB, long rB)
{
  constexpr bool AZT    = (FLAGS & GF_A_ZT) != 0;
  constexpr bool A16    = ((FLAGS & GF_A_F16) != 0) || AZT;
  constexpr bool RELU_A = (FLAGS & GF_RELU_A) != 0;
  constexpr bool RELU_O = (FLAGS & GF_RELU_O) != 0;
  constexpr bool TRANS  = (FLAGS & GF_TRANS) != 0;

  __shared__ _Float16 As[32][40];
  int tid = threadIdx.x;
  int bz  = blockIdx.z;
  const float*    Af = (const float*)Av;
  const _Float16* Ah = (const _Float16*)Av;
  long aOff = (long)bz * aB;
  float*       Cp = C + (long)bz * cB;
  const float* Rp = res ? res + (long)bz * rB : (const float*)0;
  int m0 = blockIdx.y * 32;
  int n0 = blockIdx.x * 64;
  int w  = tid >> 5, wm = w >> 2, wn = w & 3;
  const _Float16* Bp = Bt + (long)bz * bBh + (long)(n0 + wn * 16) * K;
  v8f acc = {};
  for (int k0 = 0; k0 < K; k0 += 32) {
    float av[4];
#pragma unroll
    for (int rep = 0; rep < 4; ++rep) {
      int e = tid + rep * 256;
      int r = e >> 5, c = e & 31;
      long idx;
      if (AZT) {
        // A is zT[i][c][j] f16; logical row m = i*768+j, col k -> zT index
        long iz = (long)((m0 + r) / NR);
        int  jz = (m0 + r) % NR;
        idx = (iz * CZ + (k0 + c)) * NR + jz;
      } else {
        idx = aOff + (long)(m0 + r) * K + (k0 + c);
      }
      av[rep] = A16 ? (float)Ah[idx] : Af[idx];
    }
#pragma unroll
    for (int rep = 0; rep < 4; ++rep) {
      int e = tid + rep * 256;
      int r = e >> 5, c = e & 31;
      float v = av[rep];
      if (RELU_A) v = v > 0.f ? v : 0.f;
      As[r][c] = (_Float16)v;
    }
    __syncthreads();
    v16h af = frag_ld(&As[wm * 16][0], 40);
    v16h bf = frag_ld(Bp + k0, K);
    acc = wmma_f16(af, bf, acc);
    __syncthreads();
  }
  int lane = tid & 31;
  int n  = n0 + wn * 16 + (lane & 15);
  int mb = m0 + wm * 16 + ((lane >> 4) << 3);
  if (n < N) {
    float b = bias ? bias[n] : 0.f;
#pragma unroll
    for (int r = 0; r < 8; ++r) {
      int m = mb + r;
      float v = (acc[r] + b) * out_scale;
      if (Rp) v += Rp[(long)m * N + n];
      if (RELU_O) v = v > 0.f ? v : 0.f;
      if (TRANS) Cp[(long)n * M + m] = v;
      else       Cp[(long)m * ldc + n] = v;
    }
  }
}

// ---------------------------------------------------------------------------
// LayerNorm: one block per row (128 threads), f32 out.
// ---------------------------------------------------------------------------
__global__ __launch_bounds__(128)
void layernorm_kernel(const float* __restrict__ x, const float* __restrict__ g,
                      const float* __restrict__ b, float* __restrict__ outF, int D)
{
  long row = blockIdx.x;
  const float* xr = x + row * D;
  __shared__ float r1[128], r2[128];
  float s = 0.f, s2 = 0.f;
  for (int c = threadIdx.x; c < D; c += 128) { float v = xr[c]; s += v; s2 += v * v; }
  r1[threadIdx.x] = s; r2[threadIdx.x] = s2; __syncthreads();
  for (int st = 64; st > 0; st >>= 1) {
    if (threadIdx.x < st) { r1[threadIdx.x] += r1[threadIdx.x + st]; r2[threadIdx.x] += r2[threadIdx.x + st]; }
    __syncthreads();
  }
  float mu = r1[0] / D;
  float var = r2[0] / D - mu * mu;
  float inv = rsqrtf(var + 1e-5f);
  for (int c = threadIdx.x; c < D; c += 128) {
    outF[row * D + c] = (xr[c] - mu) * inv * g[c] + b[c];
  }
}

// LayerNorm of z, written transposed f16: zT[i][c][j]  (j contiguous per (i,c))
__global__ __launch_bounds__(128)
void ln_z_kernel(const float* __restrict__ x, const float* __restrict__ g,
                 const float* __restrict__ b, _Float16* __restrict__ zT)
{
  long row = blockIdx.x;             // i*768 + j
  long i = row / NR;
  int  j = (int)(row % NR);
  const float* xr = x + row * CZ;
  __shared__ float r1[128], r2[128];
  float v0 = xr[threadIdx.x];
  r1[threadIdx.x] = v0; r2[threadIdx.x] = v0 * v0; __syncthreads();
  for (int st = 64; st > 0; st >>= 1) {
    if (threadIdx.x < st) { r1[threadIdx.x] += r1[threadIdx.x + st]; r2[threadIdx.x] += r2[threadIdx.x + st]; }
    __syncthreads();
  }
  float mu = r1[0] / CZ;
  float var = r2[0] / CZ - mu * mu;
  float inv = rsqrtf(var + 1e-5f);
  int c = threadIdx.x;
  zT[(i * CZ + c) * NR + j] = (_Float16)((v0 - mu) * inv * g[c] + b[c]);
}

// ---------------------------------------------------------------------------
// init quats/trans
// ---------------------------------------------------------------------------
__global__ void init_state_kernel(float* quats, float* trans)
{
  int i = blockIdx.x * blockDim.x + threadIdx.x;
  if (i >= NR) return;
  quats[i * 4 + 0] = 1.f; quats[i * 4 + 1] = 0.f;
  quats[i * 4 + 2] = 0.f; quats[i * 4 + 3] = 0.f;
  trans[i * 3 + 0] = 0.f; trans[i * 3 + 1] = 0.f; trans[i * 3 + 2] = 0.f;
}

// quat -> rot matrix
__global__ void quat_rot_kernel(const float* quats, float* Rb)
{
  int i = blockIdx.x * blockDim.x + threadIdx.x;
  if (i >= NR) return;
  float w = quats[i*4], x = quats[i*4+1], y = quats[i*4+2], z = quats[i*4+3];
  float* R = Rb + i * 9;
  R[0] = w*w + x*x - y*y - z*z; R[1] = 2.f*(x*y - w*z); R[2] = 2.f*(x*z + w*y);
  R[3] = 2.f*(x*y + w*z); R[4] = w*w - x*x + y*y - z*z; R[5] = 2.f*(y*z - w*x);
  R[6] = 2.f*(x*z - w*y); R[7] = 2.f*(y*z + w*x); R[8] = w*w - x*x - y*y + z*z;
}

// ---------------------------------------------------------------------------
// prep: build QE/KE (f16, K=32 per head: [q*inv48 | hw*q_pts(12) | pad4]),
// row biases qb/kb, and VextP: packed f16 V matrix [64 rows][768] per head
// (rows 0-15 = v, 16-39 = v_pts global frame, 40-63 zero pad).
// one thread per (h,i); i fastest for coalesced VextP column writes.
// ---------------------------------------------------------------------------
__global__ void prep_qkv_kernel(const float* __restrict__ P, const float* __restrict__ Rb,
                                const float* __restrict__ tr, const float* __restrict__ headw,
                                _Float16* QE, _Float16* KE, float* qb, float* kb,
                                _Float16* VextP)
{
  int idx = blockIdx.x * blockDim.x + threadIdx.x;
  if (idx >= NR * NH) return;
  int h = idx / NR, i = idx % NR;
  const float* R = Rb + i * 9;
  float tx = tr[i*3], ty = tr[i*3+1], tz = tr[i*3+2];
  float hw = log1pf(expf(headw[h])) * 0.13608276348795434f;   // sqrt(1/54)
  const float inv48 = 0.14433756729740643f;                   // 1/sqrt(48)
  const float* Pi = P + (long)i * 1152;
  _Float16* qe = QE + ((long)h * NR + i) * 32;
  _Float16* ke = KE + ((long)h * NR + i) * 32;
  _Float16* vp = VextP + (long)h * 64 * NR;   // [64][768]
#pragma unroll
  for (int c = 0; c < 16; ++c) qe[c] = (_Float16)(Pi[h * 16 + c] * inv48);
#pragma unroll
  for (int c = 0; c < 16; ++c) ke[c] = (_Float16)Pi[192 + h * 32 + c];
#pragma unroll
  for (int c = 0; c < 16; ++c) vp[(long)c * NR + i] = (_Float16)Pi[192 + h * 32 + 16 + c];
#pragma unroll
  for (int c = 40; c < 64; ++c) vp[(long)c * NR + i] = (_Float16)0.f;
  float qq = 0.f;
#pragma unroll
  for (int p = 0; p < PQ; ++p) {
    float vx = Pi[576 +  0 + h*4 + p];
    float vy = Pi[576 + 48 + h*4 + p];
    float vz = Pi[576 + 96 + h*4 + p];
    float px = R[0]*vx + R[1]*vy + R[2]*vz + tx;
    float py = R[3]*vx + R[4]*vy + R[5]*vz + ty;
    float pz = R[6]*vx + R[7]*vy + R[8]*vz + tz;
    qq += px*px + py*py + pz*pz;
    qe[16 + p*3 + 0] = (_Float16)(hw * px);
    qe[16 + p*3 + 1] = (_Float16)(hw * py);
    qe[16 + p*3 + 2] = (_Float16)(hw * pz);
  }
#pragma unroll
  for (int c = 28; c < 32; ++c) { qe[c] = (_Float16)0.f; ke[c] = (_Float16)0.f; }
  qb[h * NR + i] = -0.5f * hw * qq;
  float kk = 0.f;
#pragma unroll
  for (int p = 0; p < PQ + PV; ++p) {
    float vx = Pi[720 +   0 + h*12 + p];
    float vy = Pi[720 + 144 + h*12 + p];
    float vz = Pi[720 + 288 + h*12 + p];
    float px = R[0]*vx + R[1]*vy + R[2]*vz + tx;
    float py = R[3]*vx + R[4]*vy + R[5]*vz + ty;
    float pz = R[6]*vx + R[7]*vy + R[8]*vz + tz;
    if (p < PQ) {
      kk += px*px + py*py + pz*pz;
      ke[16 + p*3 + 0] = (_Float16)px;
      ke[16 + p*3 + 1] = (_Float16)py;
      ke[16 + p*3 + 2] = (_Float16)pz;
    } else {
      vp[(long)(16 + (p - PQ)*3 + 0) * NR + i] = (_Float16)px;
      vp[(long)(16 + (p - PQ)*3 + 1) * NR + i] = (_Float16)py;
      vp[(long)(16 + (p - PQ)*3 + 2) * NR + i] = (_Float16)pz;
    }
  }
  kb[h * NR + i] = -0.5f * hw * kk;
}

// ---------------------------------------------------------------------------
// attention logits: a[h,i,j] = QE(i)·KE(j) + bpair + qb + kb + mask term.
// One K=32 WMMA per tile, operands direct from global.
// grid = (N/64, N/32, H), block 256
// ---------------------------------------------------------------------------
__global__ __launch_bounds__(256)
void attn_logits_kernel(const _Float16* __restrict__ QE, const _Float16* __restrict__ KE,
                        const float* __restrict__ bpair, const float* __restrict__ qb,
                        const float* __restrict__ kb, const float* __restrict__ mask,
                        float* __restrict__ alog)
{
  int h = blockIdx.z;
  int i0 = blockIdx.y * 32, j0 = blockIdx.x * 64;
  int w = threadIdx.x >> 5, wm = w >> 2, wn = w & 3;
  const _Float16* qe = QE + ((long)h * NR + i0 + wm * 16) * 32;
  const _Float16* ke = KE + ((long)h * NR + j0 + wn * 16) * 32;
  v8f acc = {};
  acc = wmma_f16(frag_ld(qe, 32), frag_ld(ke, 32), acc);
  int lane = threadIdx.x & 31;
  int j  = j0 + wn * 16 + (lane & 15);
  int ib = i0 + wm * 16 + ((lane >> 4) << 3);
  float mj  = mask[j];
  float kbv = kb[h * NR + j];
#pragma unroll
  for (int r = 0; r < 8; ++r) {
    int i = ib + r;
    float v = acc[r] + bpair[((long)h * NR + i) * NR + j] + qb[h * NR + i] + kbv
            + (mask[i] * mj - 1.f) * INF_;
    alog[((long)h * NR + i) * NR + j] = v;
  }
}

// softmax over last dim (768), writes f16 probs. grid = H*N rows.
__global__ __launch_bounds__(256)
void softmax_kernel(const float* __restrict__ x, _Float16* __restrict__ y, int L)
{
  long row = blockIdx.x;
  const float* xr = x + row * L;
  __shared__ float red[256];
  float m = -1e30f;
  for (int c = threadIdx.x; c < L; c += 256) m = fmaxf(m, xr[c]);
  red[threadIdx.x] = m; __syncthreads();
  for (int s = 128; s > 0; s >>= 1) {
    if (threadIdx.x < s) red[threadIdx.x] = fmaxf(red[threadIdx.x], red[threadIdx.x + s]);
    __syncthreads();
  }
  m = red[0]; __syncthreads();
  float sum = 0.f;
  for (int c = threadIdx.x; c < L; c += 256) sum += expf(xr[c] - m);
  red[threadIdx.x] = sum; __syncthreads();
  for (int s = 128; s > 0; s >>= 1) {
    if (threadIdx.x < s) red[threadIdx.x] += red[threadIdx.x + s];
    __syncthreads();
  }
  float inv = 1.f / red[0];
  _Float16* yr = y + row * L;
  for (int c = threadIdx.x; c < L; c += 256) yr[c] = (_Float16)(expf(xr[c] - m) * inv);
}

// ---------------------------------------------------------------------------
// o_pair[i][h][c] = sum_j a[h,i,j] * z_ln[i,j,c]  -> cat cols [576, 2112)
// One block per residue i; M=12(pad16), K=768, N=128 (8 waves x 16 cols).
// zT is transposed f16 => B fragments load directly from global (L2-resident).
// ---------------------------------------------------------------------------
__global__ __launch_bounds__(256)
void opair_kernel(const _Float16* __restrict__ ah, const _Float16* __restrict__ zT,
                  float* __restrict__ cat)
{
  int i = blockIdx.x;
  __shared__ _Float16 As[16][40];
  int tid = threadIdx.x;
  int w = tid >> 5;
  const _Float16* zb = zT + (long)i * CZ * NR + (long)(w * 16) * NR;
  v8f acc = {};
  for (int k0 = 0; k0 < NR; k0 += 32) {
    _Float16 avv[2];
#pragma unroll
    for (int rep = 0; rep < 2; ++rep) {
      int e = tid + rep * 256;
      int hr = e >> 5, kk = e & 31;
      avv[rep] = (hr < NH) ? ah[((long)hr * NR + i) * NR + k0 + kk] : (_Float16)0.f;
    }
#pragma unroll
    for (int rep = 0; rep < 2; ++rep) {
      int e = tid + rep * 256;
      As[e >> 5][e & 31] = avv[rep];
    }
    __syncthreads();
    acc = wmma_f16(frag_ld(&As[0][0], 40), frag_ld(zb + k0, NR), acc);
    __syncthreads();
  }
  int lane = tid & 31;
  int cc = w * 16 + (lane & 15);
  int hb = (lane >> 4) << 3;
#pragma unroll
  for (int r = 0; r < 8; ++r) {
    int h = hb + r;
    if (h < NH) cat[(long)i * 2112 + 576 + h * CZ + cc] = acc[r];
  }
}

// ---------------------------------------------------------------------------
// post-attention: o -> cat[0,192); o_pt local frame + norms -> cat[192,576)
// one thread per (i,h).  Oext is [h][i][40] f32 (written by the batched GEMM).
// ---------------------------------------------------------------------------
__global__ void post_o_kernel(const float* __restrict__ Oext, const float* __restrict__ Rb,
                              const float* __restrict__ tr, float* __restrict__ cat)
{
  int idx = blockIdx.x * blockDim.x + threadIdx.x;
  if (idx >= NR * NH) return;
  int i = idx / NH, h = idx % NH;
  const float* oe = Oext + ((long)h * NR + i) * 40;
  const float* R  = Rb + i * 9;
  float tx = tr[i*3], ty = tr[i*3+1], tz = tr[i*3+2];
  float* ci = cat + (long)i * 2112;
#pragma unroll
  for (int c = 0; c < 16; ++c) ci[h * 16 + c] = oe[c];
#pragma unroll
  for (int p = 0; p < PV; ++p) {
    float dx = oe[16 + p*3 + 0] - tx;
    float dy = oe[16 + p*3 + 1] - ty;
    float dz = oe[16 + p*3 + 2] - tz;
    float lx = R[0]*dx + R[3]*dy + R[6]*dz;   // R^T
    float ly = R[1]*dx + R[4]*dy + R[7]*dz;
    float lz = R[2]*dx + R[5]*dy + R[8]*dz;
    float nm = sqrtf(lx*lx + ly*ly + lz*lz + 1e-8f);
    ci[192 +   0 + h*PV + p] = lx;
    ci[192 +  96 + h*PV + p] = ly;
    ci[192 + 192 + h*PV + p] = lz;
    ci[480 + h*PV + p] = nm;
  }
}

// ---------------------------------------------------------------------------
// backbone update: quats/trans update + write sm_frames for this block
// ---------------------------------------------------------------------------
__global__ void bb_update_kernel(const float* __restrict__ upd, float* quats, float* trans,
                                 const float* __restrict__ Rb, float* __restrict__ out_frames)
{
  int i = blockIdx.x * blockDim.x + threadIdx.x;
  if (i >= NR) return;
  float w = quats[i*4], x = quats[i*4+1], y = quats[i*4+2], z = quats[i*4+3];
  float b = upd[i*6], c = upd[i*6+1], d = upd[i*6+2];
  float r0 = -x*b - y*c - z*d;
  float r1 =  w*b + y*d - z*c;
  float r2 =  w*c - x*d + z*b;
  float r3 =  w*d + x*c - y*b;
  float nw = w + r0, nx = x + r1, ny = y + r2, nz = z + r3;
  float inv = rsqrtf(nw*nw + nx*nx + ny*ny + nz*nz);
  nw *= inv; nx *= inv; ny *= inv; nz *= inv;
  const float* R = Rb + i * 9;
  float ux = upd[i*6+3], uy = upd[i*6+4], uz = upd[i*6+5];
  float tx = trans[i*3+0] + R[0]*ux + R[1]*uy + R[2]*uz;
  float ty = trans[i*3+1] + R[3]*ux + R[4]*uy + R[5]*uz;
  float tz = trans[i*3+2] + R[6]*ux + R[7]*uy + R[8]*uz;
  quats[i*4] = nw; quats[i*4+1] = nx; quats[i*4+2] = ny; quats[i*4+3] = nz;
  trans[i*3] = tx; trans[i*3+1] = ty; trans[i*3+2] = tz;
  out_frames[i*7+0] = nw; out_frames[i*7+1] = nx;
  out_frames[i*7+2] = ny; out_frames[i*7+3] = nz;
  out_frames[i*7+4] = tx * SCALE_; out_frames[i*7+5] = ty * SCALE_;
  out_frames[i*7+6] = tz * SCALE_;
}

// ---------------------------------------------------------------------------
// torsion frames + atom14 positions + angle outputs. one thread per residue.
// ---------------------------------------------------------------------------
DEVFN void mm3(const float* A, const float* B, float* C) {
  for (int r = 0; r < 3; ++r)
    for (int c = 0; c < 3; ++c)
      C[r*3+c] = A[r*3]*B[c] + A[r*3+1]*B[3+c] + A[r*3+2]*B[6+c];
}
DEVFN void mv3(const float* A, const float* v, float* o) {
  for (int r = 0; r < 3; ++r) o[r] = A[r*3]*v[0] + A[r*3+1]*v[1] + A[r*3+2]*v[2];
}

__global__ void frames_kernel(const float* __restrict__ u, const float* __restrict__ trans,
                              const float* __restrict__ Rb, const int* __restrict__ aatype,
                              const float* __restrict__ def_frames, const int* __restrict__ group_idx,
                              const float* __restrict__ atom_mask, const float* __restrict__ lit_pos,
                              float* out_sc, float* out_unn, float* out_ang, float* out_pos)
{
  int i = blockIdx.x * blockDim.x + threadIdx.x;
  if (i >= NR) return;
  int aa = aatype[i];
  float als[8], alc[8];
  als[0] = 0.f; alc[0] = 1.f;
  for (int t = 0; t < 7; ++t) {
    float u0 = u[i*14 + t*2], u1 = u[i*14 + t*2 + 1];
    float dn = sqrtf(fmaxf(u0*u0 + u1*u1, 1e-8f));
    out_unn[i*14 + t*2] = u0; out_unn[i*14 + t*2 + 1] = u1;
    float s_ = u0 / dn, c_ = u1 / dn;
    out_ang[i*14 + t*2] = s_; out_ang[i*14 + t*2 + 1] = c_;
    als[t+1] = s_; alc[t+1] = c_;
  }
  float fR[8][9], ft[8][3];
  const float* df = def_frames + aa * 128;
  for (int g = 0; g < 8; ++g) {
    const float* d4 = df + g * 16;
    float c_ = alc[g], s_ = als[g];
    for (int r = 0; r < 3; ++r) {
      float d0 = d4[r*4], d1 = d4[r*4+1], d2 = d4[r*4+2];
      fR[g][r*3+0] = d0;
      fR[g][r*3+1] = d1*c_ + d2*s_;
      fR[g][r*3+2] = -d1*s_ + d2*c_;
      ft[g][r] = d4[r*4+3];
    }
  }
  float aR[8][9], at[8][3];
  for (int g = 0; g < 5; ++g) {
    for (int k = 0; k < 9; ++k) aR[g][k] = fR[g][k];
    for (int k = 0; k < 3; ++k) at[g][k] = ft[g][k];
  }
  // chained groups 5..7
  mm3(fR[4], fR[5], aR[5]); mv3(fR[4], ft[5], at[5]);
  for (int k = 0; k < 3; ++k) at[5][k] += ft[4][k];
  mm3(aR[5], fR[6], aR[6]); mv3(aR[5], ft[6], at[6]);
  for (int k = 0; k < 3; ++k) at[6][k] += at[5][k];
  mm3(aR[6], fR[7], aR[7]); mv3(aR[6], ft[7], at[7]);
  for (int k = 0; k < 3; ++k) at[7][k] += at[6][k];
  // compose with backbone
  const float* R = Rb + i * 9;
  float ts[3] = { trans[i*3] * SCALE_, trans[i*3+1] * SCALE_, trans[i*3+2] * SCALE_ };
  float gR[8][9], gt[8][3];
  for (int g = 0; g < 8; ++g) {
    mm3(R, aR[g], gR[g]); mv3(R, at[g], gt[g]);
    for (int k = 0; k < 3; ++k) gt[g][k] += ts[k];
    float* sc = out_sc + (long)i * 128 + g * 16;
    for (int r = 0; r < 3; ++r) {
      sc[r*4+0] = gR[g][r*3+0]; sc[r*4+1] = gR[g][r*3+1];
      sc[r*4+2] = gR[g][r*3+2]; sc[r*4+3] = gt[g][r];
    }
    sc[12] = 0.f; sc[13] = 0.f; sc[14] = 0.f; sc[15] = 1.f;
  }
  for (int a = 0; a < 14; ++a) {
    int grp = group_idx[aa * 14 + a];
    const float* lp = lit_pos + (aa * 14 + a) * 3;
    float p[3];
    mv3(gR[grp], lp, p);
    float m = atom_mask[aa * 14 + a];
    out_pos[i*42 + a*3 + 0] = (p[0] + gt[grp][0]) * m;
    out_pos[i*42 + a*3 + 1] = (p[1] + gt[grp][1]) * m;
    out_pos[i*42 + a*3 + 2] = (p[2] + gt[grp][2]) * m;
  }
}

// ---------------------------------------------------------------------------
// host side
// ---------------------------------------------------------------------------
static inline void gemm(hipStream_t st, const void* A, const _Float16* B,
                        const float* bias, const float* res, float* C,
                        int M, int N, int K, int flags, float scale, int ldc,
                        int batch = 1, long aB = 0, long bBh = 0, long cB = 0, long rB = 0)
{
  dim3 g((N + 63) / 64, M / 32, batch);
  switch (flags) {
    case 0:
      gemm_wmma_kernel<0><<<g, 256, 0, st>>>(A, B, bias, res, C, M, N, K, scale, ldc, aB, bBh, cB, rB); break;
    case GF_RELU_O:
      gemm_wmma_kernel<GF_RELU_O><<<g, 256, 0, st>>>(A, B, bias, res, C, M, N, K, scale, ldc, aB, bBh, cB, rB); break;
    case GF_RELU_A:
      gemm_wmma_kernel<GF_RELU_A><<<g, 256, 0, st>>>(A, B, bias, res, C, M, N, K, scale, ldc, aB, bBh, cB, rB); break;
    case GF_RELU_A | GF_RELU_O:
      gemm_wmma_kernel<GF_RELU_A | GF_RELU_O><<<g, 256, 0, st>>>(A, B, bias, res, C, M, N, K, scale, ldc, aB, bBh, cB, rB); break;
    case GF_A_F16:
      gemm_wmma_kernel<GF_A_F16><<<g, 256, 0, st>>>(A, B, bias, res, C, M, N, K, scale, ldc, aB, bBh, cB, rB); break;
    case GF_A_ZT | GF_TRANS:
      gemm_wmma_kernel<GF_A_ZT | GF_TRANS><<<g, 256, 0, st>>>(A, B, bias, res, C, M, N, K, scale, ldc, aB, bBh, cB, rB); break;
    default:
      gemm_wmma_kernel<0><<<g, 256, 0, st>>>(A, B, bias, res, C, M, N, K, scale, ldc, aB, bBh, cB, rB); break;
  }
}

extern "C" void kernel_launch(void* const* d_in, const int* in_sizes, int n_in,
                              void* d_out, int out_size, void* d_ws, size_t ws_size,
                              hipStream_t stream)
{
  (void)in_sizes; (void)n_in; (void)out_size; (void)ws_size;
  const float* s_in   = (const float*)d_in[0];
  const float* z_in   = (const float*)d_in[1];
  const float* mask   = (const float*)d_in[2];
  const int*   aatype = (const int*)d_in[3];
  const float* ln_s_g = (const float*)d_in[5];
  const float* ln_s_b = (const float*)d_in[6];
  const float* ln_z_g = (const float*)d_in[7];
  const float* ln_z_b = (const float*)d_in[8];
  const float* lin_w  = (const float*)d_in[9];
  const float* lin_b  = (const float*)d_in[10];
  const float* q_w    = (const float*)d_in[11];
  const float* q_b    = (const float*)d_in[12];
  const float* kv_w   = (const float*)d_in[13];
  const float* kv_b   = (const float*)d_in[14];
  const float* qp_w   = (const float*)d_in[15];
  const float* qp_b   = (const float*)d_in[16];
  const float* kvp_w  = (const float*)d_in[17];
  const float* kvp_b  = (const float*)d_in[18];
  const float* pb_w   = (const float*)d_in[19];
  const float* pb_b   = (const float*)d_in[20];
  const float* headw  = (const float*)d_in[21];
  const float* out_w  = (const float*)d_in[22];
  const float* out_b  = (const float*)d_in[23];
  const float* lni_g  = (const float*)d_in[24];
  const float* lni_b  = (const float*)d_in[25];
  const float* t1_w   = (const float*)d_in[26];
  const float* t1_b   = (const float*)d_in[27];
  const float* t2_w   = (const float*)d_in[28];
  const float* t2_b   = (const float*)d_in[29];
  const float* t3_w   = (const float*)d_in[30];
  const float* t3_b   = (const float*)d_in[31];
  const float* tln_g  = (const float*)d_in[32];
  const float* tln_b  = (const float*)d_in[33];
  const float* bb_w   = (const float*)d_in[34];
  const float* bb_b   = (const float*)d_in[35];
  const float* ai_w   = (const float*)d_in[36];
  const float* ai_b   = (const float*)d_in[37];
  const float* az_w   = (const float*)d_in[38];
  const float* az_b   = (const float*)d_in[39];
  const float* b0a_w  = (const float*)d_in[40];
  const float* b0a_b  = (const float*)d_in[41];
  const float* b0b_w  = (const float*)d_in[42];
  const float* b0b_b  = (const float*)d_in[43];
  const float* b1a_w  = (const float*)d_in[44];
  const float* b1a_b  = (const float*)d_in[45];
  const float* b1b_w  = (const float*)d_in[46];
  const float* b1b_b  = (const float*)d_in[47];
  const float* ao_w   = (const float*)d_in[48];
  const float* ao_b   = (const float*)d_in[49];
  const float* deff   = (const float*)d_in[50];
  const int*   gidx   = (const int*)d_in[51];
  const float* amask  = (const float*)d_in[52];
  const float* litp   = (const float*)d_in[53];

  // workspace layout
  size_t off = 0;
  char* base = (char*)d_ws;
  auto alloc = [&](size_t bytes) -> char* {
    char* p = base + off;
    off += (bytes + 255) & ~(size_t)255;
    return p;
  };
  _Float16* zT  = (_Float16*)alloc((long)NR * NR * CZ * 2);       // transposed z_ln
  _Float16* ah  = (_Float16*)alloc((long)NH * NR * NR * 2);
  float* alog   = (float*)alloc((long)NH * NR * NR * 4);
  float* bpair  = (float*)alloc((long)NH * NR * NR * 4);
  _Float16* QE  = (_Float16*)alloc((long)NH * NR * 32 * 2);
  _Float16* KE  = (_Float16*)alloc((long)NH * NR * 32 * 2);
  float* qb     = (float*)alloc(NH * NR * 4);
  float* kb     = (float*)alloc(NH * NR * 4);
  float* s_init = (float*)alloc((long)NR * CS * 4);
  float* s_cur  = (float*)alloc((long)NR * CS * 4);
  float* stmp   = (float*)alloc((long)NR * CS * 4);
  float* h1     = (float*)alloc((long)NR * CS * 4);
  float* h2     = (float*)alloc((long)NR * CS * 4);
  float* P      = (float*)alloc((long)NR * 1152 * 4);
  _Float16* VextP = (_Float16*)alloc((long)NH * 64 * NR * 2);
  float* Oext   = (float*)alloc((long)NH * NR * 40 * 4);
  float* cat    = (float*)alloc((long)NR * 2112 * 4);
  float* quats  = (float*)alloc(NR * 4 * 4);
  float* trans  = (float*)alloc(NR * 3 * 4);
  float* Rbuf   = (float*)alloc(NR * 9 * 4);
  float* upd    = (float*)alloc(NR * 6 * 4);
  float* a0     = (float*)alloc((long)NR * CANG * 4);
  float* tt1    = (float*)alloc((long)NR * CANG * 4);
  float* uang   = (float*)alloc(NR * 14 * 4);

  // packed f16 weights (transposed, Npad = ceil(N/64)*64)
  auto pack = [&](const float* W, int K, int N) -> _Float16* {
    int Npad = ((N + 63) / 64) * 64;
    _Float16* Wt = (_Float16*)alloc((long)Npad * K * 2);
    long tot = (long)Npad * K;
    pack_w_kernel<<<(int)((tot + 255) / 256), 256, 0, stream>>>(W, Wt, K, N, Npad);
    return Wt;
  };
  _Float16* lin_wp = pack(lin_w, CS, CS);
  _Float16* q_wp   = pack(q_w,  CS, NH * CH);
  _Float16* kv_wp  = pack(kv_w, CS, 2 * NH * CH);
  _Float16* qp_wp  = pack(qp_w, CS, NH * PQ * 3);
  _Float16* kvp_wp = pack(kvp_w, CS, NH * 12 * 3);
  _Float16* pb_wp  = pack(pb_w, CZ, NH);
  _Float16* out_wp = pack(out_w, 2112, CS);
  _Float16* t1_wp  = pack(t1_w, CS, CS);
  _Float16* t2_wp  = pack(t2_w, CS, CS);
  _Float16* t3_wp  = pack(t3_w, CS, CS);
  _Float16* bb_wp  = pack(bb_w, CS, 6);
  _Float16* ai_wp  = pack(ai_w, CS, CANG);
  _Float16* az_wp  = pack(az_w, CS, CANG);
  _Float16* b0a_wp = pack(b0a_w, CANG, CANG);
  _Float16* b0b_wp = pack(b0b_w, CANG, CANG);
  _Float16* b1a_wp = pack(b1a_w, CANG, CANG);
  _Float16* b1b_wp = pack(b1b_w, CANG, CANG);
  _Float16* ao_wp  = pack(ao_w, CANG, 14);

  float* outF = (float*)d_out;
  const long F_frames = 0, F_sc = 43008, F_unn = 829440, F_ang = 915456;
  const long F_pos = 1001472, F_states = 1259520, F_single = 3618816;

  // ---- pre-loop ----
  layernorm_kernel<<<NR, 128, 0, stream>>>(s_in, ln_s_g, ln_s_b, s_init, CS);
  ln_z_kernel<<<NR * NR, 128, 0, stream>>>(z_in, ln_z_g, ln_z_b, zT);
  gemm(stream, s_init, lin_wp, lin_b, 0, s_cur, NR, CS, CS, 0, 1.f, CS);
  // pair bias: sqrt(1/3)*(z_ln @ Wb + bb), A read from zT, out transposed [h][ij]
  gemm(stream, zT, pb_wp, pb_b, 0, bpair, NR * NR, NH, CZ,
       GF_A_ZT | GF_TRANS, 0.57735026918962576f, NH);
  init_state_kernel<<<3, 256, 0, stream>>>(quats, trans);

  for (int blk = 0; blk < 8; ++blk) {
    quat_rot_kernel<<<3, 256, 0, stream>>>(quats, Rbuf);
    // fused projection buffer P = [q | kv | q_pts | kv_pts]
    gemm(stream, s_cur, q_wp,   q_b,   0, P + 0,   NR, NH * CH,     CS, 0, 1.f, 1152);
    gemm(stream, s_cur, kv_wp,  kv_b,  0, P + 192, NR, 2 * NH * CH, CS, 0, 1.f, 1152);
    gemm(stream, s_cur, qp_wp,  qp_b,  0, P + 576, NR, NH * PQ * 3, CS, 0, 1.f, 1152);
    gemm(stream, s_cur, kvp_wp, kvp_b, 0, P + 720, NR, NH * 12 * 3, CS, 0, 1.f, 1152);
    prep_qkv_kernel<<<(NR * NH + 255) / 256, 256, 0, stream>>>(P, Rbuf, trans, headw,
                                                               QE, KE, qb, kb, VextP);
    {
      dim3 g(NR / 64, NR / 32, NH);
      attn_logits_kernel<<<g, 256, 0, stream>>>(QE, KE, bpair, qb, kb, mask, alog);
    }
    softmax_kernel<<<NH * NR, 256, 0, stream>>>(alog, ah, NR);
    // o / o_pt: batched per-head GEMM (A = f16 probs, B = packed VextP)
    gemm(stream, ah, VextP, 0, 0, Oext, NR, 40, NR, GF_A_F16, 1.f, 40,
         NH, (long)NR * NR, (long)64 * NR, (long)NR * 40, 0);
    opair_kernel<<<NR, 256, 0, stream>>>(ah, zT, cat);
    post_o_kernel<<<(NR * NH + 255) / 256, 256, 0, stream>>>(Oext, Rbuf, trans, cat);
    // output projection + residual, then LN
    gemm(stream, cat, out_wp, out_b, s_cur, stmp, NR, CS, 2112, 0, 1.f, CS);
    layernorm_kernel<<<NR, 128, 0, stream>>>(stmp, lni_g, lni_b, s_cur, CS);
    // transition
    gemm(stream, s_cur, t1_wp, t1_b, 0, h1, NR, CS, CS, GF_RELU_O, 1.f, CS);
    gemm(stream, h1,    t2_wp, t2_b, 0, h2, NR, CS, CS, GF_RELU_O, 1.f, CS);
    gemm(stream, h2,    t3_wp, t3_b, s_cur, stmp, NR, CS, CS, 0, 1.f, CS);
    layernorm_kernel<<<NR, 128, 0, stream>>>(stmp, tln_g, tln_b, s_cur, CS);
    // backbone update
    gemm(stream, s_cur, bb_wp, bb_b, 0, upd, NR, 6, CS, 0, 1.f, 6);
    bb_update_kernel<<<3, 256, 0, stream>>>(upd, quats, trans, Rbuf,
                                            outF + F_frames + (long)blk * NR * 7);
    quat_rot_kernel<<<3, 256, 0, stream>>>(quats, Rbuf);   // Rbb for frames
    // angle resnet
    gemm(stream, s_cur,  ai_wp, ai_b, 0,  a0, NR, CANG, CS, GF_RELU_A, 1.f, CANG);
    gemm(stream, s_init, az_wp, az_b, a0, a0, NR, CANG, CS, GF_RELU_A, 1.f, CANG);
    gemm(stream, a0,  b0a_wp, b0a_b, 0,  tt1, NR, CANG, CANG, GF_RELU_A | GF_RELU_O, 1.f, CANG);
    gemm(stream, tt1, b0b_wp, b0b_b, a0, a0,  NR, CANG, CANG, 0, 1.f, CANG);
    gemm(stream, a0,  b1a_wp, b1a_b, 0,  tt1, NR, CANG, CANG, GF_RELU_A | GF_RELU_O, 1.f, CANG);
    gemm(stream, tt1, b1b_wp, b1b_b, a0, a0,  NR, CANG, CANG, 0, 1.f, CANG);
    gemm(stream, a0,  ao_wp,  ao_b,  0,  uang, NR, 14, CANG, GF_RELU_A, 1.f, 14);
    // frames + atom positions + angle outputs
    frames_kernel<<<3, 256, 0, stream>>>(uang, trans, Rbuf, aatype, deff, gidx, amask, litp,
                                         outF + F_sc  + (long)blk * NR * 128,
                                         outF + F_unn + (long)blk * NR * 14,
                                         outF + F_ang + (long)blk * NR * 14,
                                         outF + F_pos + (long)blk * NR * 42);
    hipMemcpyAsync(outF + F_states + (long)blk * NR * CS, s_cur,
                   (size_t)NR * CS * 4, hipMemcpyDeviceToDevice, stream);
  }
  hipMemcpyAsync(outF + F_single, s_cur, (size_t)NR * CS * 4,
                 hipMemcpyDeviceToDevice, stream);
}
